// BatchTopKSAE_74534862455446
// MI455X (gfx1250) — matrix operand order repaired and checked by reading, not compiled, not run on hardware
//
#include <hip/hip_runtime.h>
#include <hip/hip_bf16.h>
#include <stdint.h>

typedef __attribute__((ext_vector_type(16))) __bf16 v16bf;
typedef __attribute__((ext_vector_type(8)))  float  v8f;

static __device__ __forceinline__ uint16_t f2bf(float f) {
  uint32_t u = __float_as_uint(f);
  u += 0x7FFFu + ((u >> 16) & 1u);        // round to nearest even
  return (uint16_t)(u >> 16);
}

// ---------------- conversion kernels ----------------
__global__ void k_conv_x(const float* __restrict__ x, const float* __restrict__ b_dec,
                         uint16_t* __restrict__ xbf, int D, long long n) {
  long long i = (long long)blockIdx.x * blockDim.x + threadIdx.x;
  if (i < n) xbf[i] = f2bf(x[i] - b_dec[(int)(i % D)]);
}

__global__ void k_conv_w(const float* __restrict__ w, uint16_t* __restrict__ wbf, long long n) {
  long long i = (long long)blockIdx.x * blockDim.x + threadIdx.x;
  if (i < n) wbf[i] = f2bf(w[i]);
}

__global__ void k_zero_u32(unsigned* __restrict__ p, int n) {
  int i = blockIdx.x * blockDim.x + threadIdx.x;
  if (i < n) p[i] = 0u;
}

// ---------------- encoder GEMM: acts = relu((x-b_dec) @ W_enc^T + b_enc) ----------------
// 256 threads = 8 waves; block tile 128(M) x 128(N); wave grid 4(M) x 2(N),
// each wave computes 32x64 = 2x4 tiles of v_wmma_f32_16x16x32_bf16.
// A/B K-slices (128x32 bf16 each) staged in LDS with double-buffered
// global_load_async_to_lds_b128 (ASYNCcnt), fragments read back with ds_load_b128.
#define ROWPAD 40   // LDS row stride in elements (80 bytes: 16B-aligned, conflict-free)

__global__ __launch_bounds__(256)
void k_gemm_enc(const uint16_t* __restrict__ xbf, const uint16_t* __restrict__ wbf,
                const float* __restrict__ b_enc, float* __restrict__ acts,
                int D, int F) {
  __shared__ __align__(16) uint16_t smA[2][128 * ROWPAD];
  __shared__ __align__(16) uint16_t smB[2][128 * ROWPAD];

  const int tid  = threadIdx.x;
  const int lane = tid & 31;
  const int wave = tid >> 5;
  const int wm   = wave & 3;                 // wave position along M (0..3)
  const int wn   = wave >> 2;                // wave position along N (0..1)
  const int lm   = lane & 15;
  const int hi   = lane >> 4;
  const int mBlk = blockIdx.y * 128;
  const int fBlk = blockIdx.x * 128;

  // --- staging assignment: each thread async-copies 32B of one A row + one B row ---
  const int rr   = tid >> 1;                 // row 0..127
  const int boff = (tid & 1) * 32;           // byte offset within the 64B row slice
  const uint16_t* gA = xbf + (size_t)(mBlk + rr) * D + (boff >> 1);
  const uint16_t* gB = wbf + (size_t)(fBlk + rr) * D + (boff >> 1);
  unsigned lA[2], lB[2];
#pragma unroll
  for (int s = 0; s < 2; ++s) {
    lA[s] = (unsigned)(uintptr_t)((const char*)&smA[s][rr * ROWPAD] + boff);
    lB[s] = (unsigned)(uintptr_t)((const char*)&smB[s][rr * ROWPAD] + boff);
  }

  // --- per-wave fragment base pointers in LDS ---
  const uint16_t* aP[2][2];
  const uint16_t* bP[2][4];
#pragma unroll
  for (int s = 0; s < 2; ++s) {
#pragma unroll
    for (int i = 0; i < 2; ++i) aP[s][i] = &smA[s][(wm * 32 + i * 16 + lm) * ROWPAD];
#pragma unroll
    for (int j = 0; j < 4; ++j) bP[s][j] = &smB[s][(wn * 64 + j * 16 + lm) * ROWPAD];
  }

  v8f c[2][4] = {};

  const int NK = D / 32;

  // prologue: fill buffer 0  (offset: applies to BOTH lds and global addr -> +16B chunk)
  {
    uint64_t a = (uint64_t)(uintptr_t)gA, b = (uint64_t)(uintptr_t)gB;
    asm volatile("global_load_async_to_lds_b128 %0, %1, off"           :: "v"(lA[0]), "v"(a) : "memory");
    asm volatile("global_load_async_to_lds_b128 %0, %1, off offset:16" :: "v"(lA[0]), "v"(a) : "memory");
    asm volatile("global_load_async_to_lds_b128 %0, %1, off"           :: "v"(lB[0]), "v"(b) : "memory");
    asm volatile("global_load_async_to_lds_b128 %0, %1, off offset:16" :: "v"(lB[0]), "v"(b) : "memory");
  }

  for (int kt = 0; kt < NK; ++kt) {
    asm volatile("s_wait_asynccnt 0" ::: "memory");   // own async writes to LDS done
    __syncthreads();                                   // everyone's writes visible; prev reads done
    const int buf = kt & 1;

    if (kt + 1 < NK) {                                 // prefetch next K-slice into other buffer
      uint64_t a = (uint64_t)(uintptr_t)(gA + (kt + 1) * 32);
      uint64_t b = (uint64_t)(uintptr_t)(gB + (kt + 1) * 32);
      const unsigned la = lA[buf ^ 1], lb = lB[buf ^ 1];
      asm volatile("global_load_async_to_lds_b128 %0, %1, off"           :: "v"(la), "v"(a) : "memory");
      asm volatile("global_load_async_to_lds_b128 %0, %1, off offset:16" :: "v"(la), "v"(a) : "memory");
      asm volatile("global_load_async_to_lds_b128 %0, %1, off"           :: "v"(lb), "v"(b) : "memory");
      asm volatile("global_load_async_to_lds_b128 %0, %1, off offset:16" :: "v"(lb), "v"(b) : "memory");
    }

    // fragments from LDS (ds_load_b128), then 8 WMMAs
    union { uint4 q[2]; v16bf v; } A[2], Bf[4];
#pragma unroll
    for (int i = 0; i < 2; ++i) {
      A[i].q[0] = *(const uint4*)(aP[buf][i] + hi * 8);
      A[i].q[1] = *(const uint4*)(aP[buf][i] + 16 + hi * 8);
    }
#pragma unroll
    for (int j = 0; j < 4; ++j) {
      Bf[j].q[0] = *(const uint4*)(bP[buf][j] + hi * 8);
      Bf[j].q[1] = *(const uint4*)(bP[buf][j] + 16 + hi * 8);
    }
#pragma unroll
    for (int j = 0; j < 4; ++j) {
      c[0][j] = __builtin_amdgcn_wmma_f32_16x16x32_bf16(false, A[0].v, false, Bf[j].v,
                                                        (short)0, c[0][j], false, false);
      c[1][j] = __builtin_amdgcn_wmma_f32_16x16x32_bf16(false, A[1].v, false, Bf[j].v,
                                                        (short)0, c[1][j], false, false);
    }
  }

  // Epilogue: + b_enc, ReLU, store.  D layout: VGPR r -> (M = r + hi*8, N = lm)
#pragma unroll
  for (int j = 0; j < 4; ++j) {
    const int f = fBlk + wn * 64 + j * 16 + lm;
    const float be = b_enc[f];
#pragma unroll
    for (int i = 0; i < 2; ++i) {
      const int mb = mBlk + wm * 32 + i * 16 + hi * 8;
#pragma unroll
      for (int r = 0; r < 8; ++r) {
        float v = c[i][j][r] + be;
        v = v > 0.f ? v : 0.f;
        acts[(size_t)(mb + r) * F + f] = v;
      }
    }
  }
}

// ---------------- batch top-k threshold via 2-level radix histogram ----------------
__global__ void k_hist1(const float* __restrict__ acts, unsigned* __restrict__ hist1, long long n) {
  __shared__ unsigned lh[8192];
  for (int i = threadIdx.x; i < 8192; i += blockDim.x) lh[i] = 0u;
  __syncthreads();
  long long stride = (long long)gridDim.x * blockDim.x;
  for (long long i = (long long)blockIdx.x * blockDim.x + threadIdx.x; i < n; i += stride) {
    float v = acts[i];
    if (v > 0.f) atomicAdd(&lh[__float_as_uint(v) >> 18], 1u);
  }
  __syncthreads();
  for (int i = threadIdx.x; i < 8192; i += blockDim.x) {
    unsigned cnt = lh[i];
    if (cnt) atomicAdd(&hist1[i], cnt);
  }
}

__global__ void k_scan1(const unsigned* __restrict__ hist1, unsigned* __restrict__ scal,
                        const int* __restrict__ kptr, int B) {
  unsigned K = (unsigned)kptr[0] * (unsigned)B;
  unsigned cum = 0, b1 = 0, above = 0;
  for (int i = 8191; i >= 1; --i) {
    unsigned h = hist1[i];
    if (cum + h >= K) { b1 = (unsigned)i; above = cum; goto done; }
    cum += h;
  }
  b1 = 0; above = cum;
done:
  scal[0] = b1;
  scal[1] = above;
}

__global__ void k_hist2(const float* __restrict__ acts, const unsigned* __restrict__ scal,
                        unsigned* __restrict__ hist2, long long n) {
  const unsigned b1 = scal[0];
  long long stride = (long long)gridDim.x * blockDim.x;
  for (long long i = (long long)blockIdx.x * blockDim.x + threadIdx.x; i < n; i += stride) {
    float v = acts[i];
    if (v > 0.f) {
      unsigned u = __float_as_uint(v);
      if ((u >> 18) == b1) atomicAdd(&hist2[(u >> 2) & 0xFFFFu], 1u);
    }
  }
}

__global__ void k_scan2(const unsigned* __restrict__ hist2, unsigned* __restrict__ scal,
                        const int* __restrict__ kptr, int B) {
  unsigned K = (unsigned)kptr[0] * (unsigned)B;
  unsigned above = scal[1];
  unsigned K2 = (K > above) ? (K - above) : 0u;
  unsigned cum = 0, b2 = 0;
  for (int i = 65535; i >= 1; --i) {
    unsigned h = hist2[i];
    if (cum + h >= K2) { b2 = (unsigned)i; break; }
    cum += h;
  }
  scal[2] = (scal[0] << 18) | (b2 << 2);   // threshold bit pattern T
}

// ---------------- sparse decode: x_hat[b,:] = b_dec + sum_sel val * W_enc[f,:] ----------------
#define DEC_CAP 2048
__global__ __launch_bounds__(256)
void k_decode(const float* __restrict__ acts, const float* __restrict__ wenc,
              const float* __restrict__ b_dec, const unsigned* __restrict__ scal,
              float* __restrict__ out, int D, int F) {
  __shared__ int   counts[257];
  __shared__ int   fidx[DEC_CAP];
  __shared__ float fval[DEC_CAP];
  const int b   = blockIdx.x;
  const int tid = threadIdx.x;
  const unsigned T = scal[2];
  const float* row = acts + (size_t)b * F;
  const int per = F >> 8;
  const int lo  = tid * per;

  int cnt = 0;
  for (int t = 0; t < per; ++t) {
    float v = row[lo + t];
    if (v > 0.f && __float_as_uint(v) >= T) ++cnt;
  }
  counts[tid] = cnt;
  __syncthreads();
  if (tid == 0) {                           // deterministic exclusive prefix scan
    int s = 0;
    for (int i = 0; i < 256; ++i) { int c2 = counts[i]; counts[i] = s; s += c2; }
    counts[256] = s;
  }
  __syncthreads();
  int pos = counts[tid];
  for (int t = 0; t < per; ++t) {
    float v = row[lo + t];
    if (v > 0.f && __float_as_uint(v) >= T) {
      if (pos < DEC_CAP) { fidx[pos] = lo + t; fval[pos] = v; }
      ++pos;
    }
  }
  __syncthreads();
  int nnz = counts[256];
  if (nnz > DEC_CAP) nnz = DEC_CAP;

  for (int j = 0; j < (D >> 8); ++j) {
    const int d = tid + (j << 8);
    float acc = b_dec[d];
    for (int i = 0; i < nnz; ++i)
      acc = fmaf(fval[i], wenc[(size_t)fidx[i] * D + d], acc);   // W_enc row f == W_dec column f
    out[(size_t)b * D + d] = acc;
  }
}

// ---------------- launcher ----------------
extern "C" void kernel_launch(void* const* d_in, const int* in_sizes, int n_in,
                              void* d_out, int out_size, void* d_ws, size_t ws_size,
                              hipStream_t stream) {
  const float* x     = (const float*)d_in[0];
  const float* W_enc = (const float*)d_in[1];
  const float* b_enc = (const float*)d_in[2];
  const float* b_dec = (const float*)d_in[4];
  const int*   kptr  = (const int*)d_in[5];

  const int D = in_sizes[4];          // 2048
  const int F = in_sizes[2];          // 32768
  const long long BD = in_sizes[0];
  const int B = (int)(BD / D);        // 4096
  const long long BF = (long long)B * F;
  const long long FD = (long long)F * D;

  char* ws = (char*)d_ws;
  size_t off = 0;
  auto alloc = [&](size_t bytes) { size_t p = off; off = (off + bytes + 255) & ~(size_t)255; return p; };
  uint16_t* xbf   = (uint16_t*)(ws + alloc((size_t)BD * 2));
  uint16_t* wbf   = (uint16_t*)(ws + alloc((size_t)FD * 2));
  float*    acts  = (float*)   (ws + alloc((size_t)BF * 4));
  unsigned* hist1 = (unsigned*)(ws + alloc(8192 * 4));
  unsigned* hist2 = (unsigned*)(ws + alloc(65536 * 4));
  unsigned* scal  = (unsigned*)(ws + alloc(4 * 4));
  float*    out   = (float*)d_out;

  // zero histogram region (hist1..scal are consecutive allocations)
  {
    int span_u32 = (int)(((char*)scal + 16) - (char*)hist1) / 4;
    k_zero_u32<<<(span_u32 + 255) / 256, 256, 0, stream>>>(hist1, span_u32);
  }

  k_conv_x<<<(int)((BD + 255) / 256), 256, 0, stream>>>(x, b_dec, xbf, D, BD);
  k_conv_w<<<(int)((FD + 255) / 256), 256, 0, stream>>>(W_enc, wbf, FD);

  dim3 ggrid(F / 128, B / 128);
  k_gemm_enc<<<ggrid, 256, 0, stream>>>(xbf, wbf, b_enc, acts, D, F);

  k_hist1<<<1024, 256, 0, stream>>>(acts, hist1, BF);
  k_scan1<<<1, 1, 0, stream>>>(hist1, scal, kptr, B);
  k_hist2<<<1024, 256, 0, stream>>>(acts, scal, hist2, BF);
  k_scan2<<<1, 1, 0, stream>>>(hist2, scal, kptr, B);

  k_decode<<<B, 256, 0, stream>>>(acts, W_enc, b_dec, scal, out, D, F);
}
